// TConv_18485539242514
// MI455X (gfx1250) — matrix operand order repaired and checked by reading, not compiled, not run on hardware
//
#include <hip/hip_runtime.h>
#include <hip/hip_bf16.h>
#include <math.h>

// ---------------- problem constants ----------------
#define Bsz   16
#define CIN   256
#define COUT  256
#define DIMIN 32
#define KK    3
#define Hh    128
#define Ww    128
#define HW    (Hh*Ww)          // 16384
#define KTOT  (CIN*KK*KK)      // 2304
#define ATTW  (COUT + CIN + KK*KK) // 521

// ---------------- workspace layout (bytes) ----------------
#define OFF_BX   ((size_t)0)                               // bf16 base_x: 128MB
#define OFF_EW   ((size_t)Bsz*CIN*HW*2)                    // bf16 eff_w (K-reordered): 18MB
#define OFF_INX  (OFF_EW + (size_t)Bsz*COUT*KTOT*2)        // in_x: B*CIN float
#define OFF_ATT  (OFF_INX + (size_t)Bsz*CIN*4)             // att_raw: B*521 float

typedef __attribute__((ext_vector_type(16))) __bf16 v16bf;
typedef __attribute__((ext_vector_type(8)))  float  v8f;
typedef __attribute__((ext_vector_type(4)))  int    v4i;

#if defined(__gfx1250__) && \
    __has_builtin(__builtin_amdgcn_global_load_async_to_lds_b128) && \
    __has_builtin(__builtin_amdgcn_s_wait_asynccnt)
#define USE_ASYNC 1
#else
#define USE_ASYNC 0
#endif

__device__ __forceinline__ unsigned short f2bf(float f) {
    union { float f; unsigned u; } v; v.f = f;
    unsigned r = v.u + 0x7FFFu + ((v.u >> 16) & 1u);   // round-to-nearest-even
    return (unsigned short)(r >> 16);
}

// ---------------------------------------------------------------
// K1: per-(b,c) plane mean  +  bf16 conversion of base_x
// ---------------------------------------------------------------
__global__ __launch_bounds__(256) void k_reduce_convert(
    const float* __restrict__ base_x, unsigned short* __restrict__ bx16,
    float* __restrict__ in_x)
{
    const int plane = blockIdx.x;              // b*CIN + c
    const float4* __restrict__ src = (const float4*)(base_x + (size_t)plane * HW);
    uint2* __restrict__ dst = (uint2*)(bx16 + (size_t)plane * HW);
    const int tid = threadIdx.x;
    float s = 0.f;
#pragma unroll
    for (int it = 0; it < 16; ++it) {
        int idx = tid + it * 256;              // 4096 float4 per plane
        float4 v = src[idx];
        s += v.x + v.y + v.z + v.w;
        uint2 p;
        p.x = (unsigned)f2bf(v.x) | ((unsigned)f2bf(v.y) << 16);
        p.y = (unsigned)f2bf(v.z) | ((unsigned)f2bf(v.w) << 16);
        dst[idx] = p;
    }
    __shared__ float red[256];
    red[tid] = s;
    __syncthreads();
    for (int off = 128; off > 0; off >>= 1) {
        if (tid < off) red[tid] += red[tid + off];
        __syncthreads();
    }
    if (tid == 0) in_x[plane] = red[0] * (1.0f / (float)HW);
}

// ---------------------------------------------------------------
// K2: 3-layer MLP (trivial size), one block of 256 threads
// ---------------------------------------------------------------
__global__ __launch_bounds__(256) void k_mlp(
    const float* __restrict__ tran_x,
    const float* __restrict__ in_x,
    const float* __restrict__ w1, const float* __restrict__ b1,
    const float* __restrict__ w2, const float* __restrict__ b2,
    const float* __restrict__ w3, const float* __restrict__ b3,
    float* __restrict__ att_raw)
{
    __shared__ float hs[Bsz][DIMIN + CIN];   // 16x288
    __shared__ float h1s[Bsz][COUT];         // 16x256
    __shared__ float h2s[Bsz][16];           // 16x16
    const int tid = threadIdx.x;

    for (int idx = tid; idx < Bsz * (DIMIN + CIN); idx += 256) {
        int b = idx / (DIMIN + CIN);
        int j = idx - b * (DIMIN + CIN);
        hs[b][j] = (j < DIMIN) ? tran_x[b * DIMIN + j]
                               : in_x[b * CIN + (j - DIMIN)];
    }
    __syncthreads();

    {   // layer 1: thread t owns output column o=t
        const float* wr = w1 + (size_t)tid * (DIMIN + CIN);
#pragma unroll 1
        for (int b = 0; b < Bsz; ++b) {
            float acc = b1[tid];
            for (int j = 0; j < DIMIN + CIN; ++j) acc += hs[b][j] * wr[j];
            h1s[b][tid] = fmaxf(acc, 0.f);
        }
    }
    __syncthreads();

    {   // layer 2
        int b = tid >> 4, o = tid & 15;
        const float* wr = w2 + (size_t)o * COUT;
        float acc = b2[o];
        for (int j = 0; j < COUT; ++j) acc += h1s[b][j] * wr[j];
        h2s[b][o] = fmaxf(acc, 0.f);
    }
    __syncthreads();

    // layer 3
    for (int idx = tid; idx < Bsz * ATTW; idx += 256) {
        int b = idx / ATTW;
        int o = idx - b * ATTW;
        const float* wr = w3 + (size_t)o * 16;
        float acc = b3[o];
#pragma unroll
        for (int j = 0; j < 16; ++j) acc += h2s[b][j] * wr[j];
        att_raw[idx] = acc;
    }
}

// ---------------------------------------------------------------
// K3: eff_w, K-REORDERED layout: eff_w[b][o][ k*256 + i ],  k = dy*3+dx
// grid = (COUT, B), 256 threads (thread = i)
// ---------------------------------------------------------------
__global__ __launch_bounds__(256) void k_effw(
    const float* __restrict__ att_raw, const float* __restrict__ conv_w,
    unsigned short* __restrict__ eff_w)
{
    const int o = blockIdx.x;
    const int b = blockIdx.y;
    const int i = threadIdx.x;
    __shared__ float w3s[KK * KK];
    if (i < KK * KK) w3s[i] = att_raw[b * ATTW + COUT + CIN + i];
    __syncthreads();

    // reshape semantics: scale uses w1a indexed by i, w2a indexed by o
    const float w1a = att_raw[b * ATTW + i];
    const float w2a = att_raw[b * ATTW + COUT + o];
    const float s = 1.0f / (1.0f + __expf(-(w1a + w2a)));

    float e[KK * KK];
    float m = -1e30f;
#pragma unroll
    for (int k = 0; k < KK * KK; ++k) { e[k] = s * w3s[k]; m = fmaxf(m, e[k]); }
    float den = 0.f;
#pragma unroll
    for (int k = 0; k < KK * KK; ++k) { e[k] = __expf(e[k] - m); den += e[k]; }
    const float scale = (float)(KK * KK) / den;

    const float* cw = conv_w + ((size_t)o * CIN + i) * (KK * KK);
    unsigned short* dst = eff_w + ((size_t)(b * COUT + o)) * KTOT + i;
#pragma unroll
    for (int k = 0; k < KK * KK; ++k)
        dst[k * CIN] = f2bf(cw[k] * e[k] * scale);   // coalesced per-k across i
}

// ---------------------------------------------------------------
// K4: implicit-GEMM conv via v_wmma_f32_16x16x32_bf16, double-buffered LDS,
//     async global->LDS copies for the A stream when available.
//     per batch: out(256 x 16384) = eff_w(256 x 2304) * im2col(2304 x 16384)
// grid = (HW/64, COUT/128, B), block = 256 threads (8 waves)
// block tile 128(M) x 64(N); wave = 32x32 sub-tile -> 4 WMMAs / K-step
// ---------------------------------------------------------------
#define LDP 40   // padded LDS row stride in halfs (80B: 16B-aligned rows, no bank hotspots)

__global__ __launch_bounds__(256) void k_conv_wmma(
    const unsigned short* __restrict__ bx16,
    const unsigned short* __restrict__ eff_w,
    const float* __restrict__ conv_b,
    float* __restrict__ out)
{
    const int b  = blockIdx.z;
    const int m0 = blockIdx.y * 128;        // cout tile base
    const int n0 = blockIdx.x * 64;         // spatial tile base (single image row)
    const int y  = n0 >> 7;
    const int x0 = n0 & 127;

    __shared__ unsigned short As[2][128 * LDP];  // [m][k] 128x32 used
    __shared__ unsigned short Bs[2][64 * LDP];   // [n][k] 64x32 used

    const int tid  = threadIdx.x;
    const int wave = tid >> 5;
    const int lane = tid & 31;
    const int mq   = wave >> 1;             // 0..3 : 32-row strip
    const int nq   = wave & 1;              // 0..1 : 32-col strip

    v8f acc00 = {}, acc01 = {}, acc10 = {}, acc11 = {};

    const unsigned short* ewB = eff_w + (size_t)(b * COUT + m0) * KTOT;
    const unsigned short* bxB = bx16 + (size_t)b * CIN * HW;

    // ---- loop-invariant loader indices
    const int a_m  = tid >> 2;              // A loader: row base (4 x b128 per 32-half row)
    const int a_q  = tid & 3;               //   which 16B chunk in the row
    const int b_n  = tid & 63;              // B gather: column (invariant over it)
    const int b_kk = tid >> 6;              //   kk base (per-it +4)
    const int gx0  = x0 + b_n - 1;

    // ---- loop-invariant fragment gather indices
    const int am0   = mq * 32 + (lane & 15);
    const int akoff = (lane < 16) ? 0 : 8;
    const int bn0   = nq * 32 + (lane & 15);
    const int bkoff = (lane < 16) ? 0 : 8;

    // tile loader: stage 32 K-values starting at kbase into buffer p
    auto loadTile = [&](int kbase, int p) {
        // A: 128 x 32 halfs, 16B-granule stream (global 16B-aligned: KTOT*2=4608B rows)
        {
            const unsigned short* srcBase = ewB + kbase;
#pragma unroll
            for (int it = 0; it < 2; ++it) {
                int m = a_m + it * 64;
                const unsigned short* g = srcBase + (size_t)m * KTOT + a_q * 8;
                unsigned short* l = &As[p][m * LDP + a_q * 8];
#if USE_ASYNC
                __builtin_amdgcn_global_load_async_to_lds_b128(
                    (__attribute__((address_space(1))) v4i*)(void*)g,
                    (__attribute__((address_space(3))) v4i*)(void*)l,
                    0, 0);
#else
                *(uint4*)l = *(const uint4*)g;
#endif
            }
        }
        // B: im2col gather; (dy,dx) uniform over the 32-wide K tile
        {
            int r9 = kbase >> 8;            // kbase / 256
            int cb = kbase & 255;
            int dy = (r9 * 11) >> 5;        // r9/3 for r9 in [0,8]
            int dx = r9 - dy * 3;
            int gy = y + dy - 1;
            int gx = gx0 + dx;
            bool ok = ((unsigned)gx < (unsigned)Ww) && ((unsigned)gy < (unsigned)Hh);
            int rowoff = (gy << 7) + gx;
#pragma unroll
            for (int it = 0; it < 8; ++it) {
                int kk = b_kk + it * 4;
                int c  = cb + kk;
                unsigned short v = 0;
                if (ok) v = bxB[((size_t)c << 14) + rowoff];
                Bs[p][b_n * LDP + kk] = v;
            }
        }
    };

    // publish a buffer: own async copies done, then block-wide barrier
    auto publish = [&]() {
#if USE_ASYNC
        __builtin_amdgcn_s_wait_asynccnt(0);
#endif
        __syncthreads();
    };

    // compute 4 WMMAs from buffer p (all b128 LDS loads, 16B-aligned)
    auto compute = [&](int p) {
        const unsigned short* ap0 = &As[p][am0 * LDP];
        const unsigned short* ap1 = ap0 + 16 * LDP;
        const unsigned short* bp0 = &Bs[p][bn0 * LDP + bkoff];
        const unsigned short* bp1 = bp0 + 16 * LDP;
        union Frag { uint4 q[2]; v16bf v; } fa0, fa1, fb0, fb1;
        // A 16x32 bf16: lanes<16 -> K {0..7,16..23}; lanes>=16 -> K {8..15,24..31}
        fa0.q[0] = *(const uint4*)(ap0 + akoff);
        fa0.q[1] = *(const uint4*)(ap0 + 16 + akoff);
        fa1.q[0] = *(const uint4*)(ap1 + akoff);
        fa1.q[1] = *(const uint4*)(ap1 + 16 + akoff);
        // B 32x16 bf16: lane = column, lanes<16 -> K {0..7,16..23}; lanes>=16 -> K {8..15,24..31}
        fb0.q[0] = *(const uint4*)(bp0 + 0);
        fb0.q[1] = *(const uint4*)(bp0 + 16);
        fb1.q[0] = *(const uint4*)(bp1 + 0);
        fb1.q[1] = *(const uint4*)(bp1 + 16);

        acc00 = __builtin_amdgcn_wmma_f32_16x16x32_bf16(
                    false, fa0.v, false, fb0.v, (short)0, acc00, false, false);
        acc01 = __builtin_amdgcn_wmma_f32_16x16x32_bf16(
                    false, fa0.v, false, fb1.v, (short)0, acc01, false, false);
        acc10 = __builtin_amdgcn_wmma_f32_16x16x32_bf16(
                    false, fa1.v, false, fb0.v, (short)0, acc10, false, false);
        acc11 = __builtin_amdgcn_wmma_f32_16x16x32_bf16(
                    false, fa1.v, false, fb1.v, (short)0, acc11, false, false);
    };

    // ---- double-buffered main loop: 36 iterations of 2 K-steps
    loadTile(0, 0);
#pragma unroll 1
    for (int k0 = 0; k0 < KTOT; k0 += 64) {
        publish();                          // buf0 ready; buf1 reads done
        loadTile(k0 + 32, 1);
        compute(0);
        publish();                          // buf1 ready; buf0 reads done
        if (k0 + 64 < KTOT) loadTile(k0 + 64, 0);
        compute(1);
    }

    // ---- epilogue: D layout — VGPR r, lane L: N = L&15, M = r + 8*(L>=16)
    const int nc   = lane & 15;
    const int mAdd = (lane >> 4) << 3;
#pragma unroll
    for (int r = 0; r < 8; ++r) {
        int o0 = m0 + mq * 32 + r + mAdd;
        int o1 = o0 + 16;
        float bias0 = conv_b[o0];
        float bias1 = conv_b[o1];
        size_t row0 = (((size_t)b * COUT + o0) << 14);
        size_t row1 = (((size_t)b * COUT + o1) << 14);
        int nA = n0 + nq * 32 + nc;
        int nB = nA + 16;
        out[row0 + nA] = acc00[r] + bias0;
        out[row0 + nB] = acc01[r] + bias0;
        out[row1 + nA] = acc10[r] + bias1;
        out[row1 + nB] = acc11[r] + bias1;
    }
}

// ---------------------------------------------------------------
extern "C" void kernel_launch(void* const* d_in, const int* in_sizes, int n_in,
                              void* d_out, int out_size, void* d_ws, size_t ws_size,
                              hipStream_t stream) {
    const float* base_x = (const float*)d_in[0];
    const float* tran_x = (const float*)d_in[1];
    const float* w1     = (const float*)d_in[2];
    const float* b1     = (const float*)d_in[3];
    const float* w2     = (const float*)d_in[4];
    const float* b2     = (const float*)d_in[5];
    const float* w3     = (const float*)d_in[6];
    const float* b3     = (const float*)d_in[7];
    const float* conv_w = (const float*)d_in[8];
    const float* conv_b = (const float*)d_in[9];
    float* out = (float*)d_out;

    char* ws = (char*)d_ws;
    unsigned short* bx16    = (unsigned short*)(ws + OFF_BX);
    unsigned short* eff_w   = (unsigned short*)(ws + OFF_EW);
    float*          in_x    = (float*)(ws + OFF_INX);
    float*          att_raw = (float*)(ws + OFF_ATT);

    k_reduce_convert<<<Bsz * CIN, 256, 0, stream>>>(base_x, bx16, in_x);
    k_mlp<<<1, 256, 0, stream>>>(tran_x, in_x, w1, b1, w2, b2, w3, b3, att_raw);
    k_effw<<<dim3(COUT, Bsz), 256, 0, stream>>>(att_raw, conv_w, eff_w);
    k_conv_wmma<<<dim3(HW / 64, COUT / 128, Bsz), 256, 0, stream>>>(
        bx16, eff_w, conv_b, out);
}
